// Attn_77584289235288
// MI455X (gfx1250) — matrix operand order repaired and checked by reading, not compiled, not run on hardware
//
#include <hip/hip_runtime.h>

#define NPTS   32768
#define NH     8
#define HD     24
#define BLK    128
#define NB     (NPTS / BLK)
#define NR     3
#define HASHD  28   /* padded to 32 in storage */

typedef __bf16 bf16_t;
typedef bf16_t v16bf __attribute__((ext_vector_type(16)));
typedef float  v8f   __attribute__((ext_vector_type(8)));

struct alignas(16) U4 { unsigned x, y, z, w; };

__device__ __forceinline__ unsigned short f2bf(float f) {
  unsigned u = __float_as_uint(f);
  unsigned r = ((u >> 16) & 1u) + 0x7FFFu;      // round-to-nearest-even
  return (unsigned short)((u + r) >> 16);
}
__device__ __forceinline__ float bf2f(unsigned short s) {
  return __uint_as_float(((unsigned)s) << 16);
}

/* Async global -> LDS (16B), tracked by ASYNCcnt. Per-lane LDS offset comes
   from the low 32 bits of the generic shared pointer (flat aperture mapping:
   low 32 bits give the LDS address). */
__device__ __forceinline__ void async_g2l_b128(void* lds_dst, const void* gsrc) {
  unsigned l = (unsigned)(unsigned long long)lds_dst;
  asm volatile("global_load_async_to_lds_b128 %0, %1, off"
               :: "v"(l), "v"(gsrc) : "memory");
}
__device__ __forceinline__ void wait_async0() {
  asm volatile("s_wait_asynccnt 0x0" ::: "memory");
}

/* ---------------- LayerNorm 1 ---------------- */
__global__ void ln1_kernel(const float* __restrict__ x, const float* __restrict__ g,
                           const float* __restrict__ b, float* __restrict__ xn) {
  int n = blockIdx.x * blockDim.x + threadIdx.x;
  if (n >= NPTS) return;
  const float* r = x + (size_t)n * HD;
  float mu = 0.f;
  #pragma unroll
  for (int i = 0; i < HD; ++i) mu += r[i];
  mu *= (1.0f / HD);
  float var = 0.f;
  #pragma unroll
  for (int i = 0; i < HD; ++i) { float d = r[i] - mu; var += d * d; }
  var *= (1.0f / HD);
  float rs = rsqrtf(var + 1e-5f);
  #pragma unroll
  for (int i = 0; i < HD; ++i)
    xn[(size_t)n * HD + i] = (r[i] - mu) * rs * g[i] + b[i];
}

/* ---------------- Q/K/V projection: xn(N,24) @ W(24,192) ---------------- */
__global__ void qkv_kernel(const float* __restrict__ xn,
                           const float* __restrict__ Wq, const float* __restrict__ Wk,
                           const float* __restrict__ Wv,
                           float* __restrict__ q, float* __restrict__ k, float* __restrict__ v) {
  __shared__ float xr[HD];
  int n = blockIdx.x, j = threadIdx.x;
  if (j < HD) xr[j] = xn[(size_t)n * HD + j];
  __syncthreads();
  float aq = 0.f, ak = 0.f, av = 0.f;
  #pragma unroll
  for (int i = 0; i < HD; ++i) {
    float xv = xr[i];
    aq += xv * Wq[i * 192 + j];
    ak += xv * Wk[i * 192 + j];
    av += xv * Wv[i * 192 + j];
  }
  size_t o = (size_t)n * 192 + j;
  q[o] = aq; k[o] = ak; v[o] = av;
}

/* ---------------- RPE weights: w[h][c] = mean(Wm[h,:,c,:])^2 ---------------- */
__global__ void wrpe_kernel(const float* __restrict__ W, float* __restrict__ wbuf) {
  int t = threadIdx.x;
  if (t >= 16) return;
  int h = t >> 1, c = t & 1;
  float s = 0.f;
  for (int d = 0; d < HD; ++d)
    for (int j = 0; j < 8; ++j)
      s += W[(h * HD + d) * 16 + c * 8 + j];
  s *= (1.0f / 192.0f);
  float w = s * s;
  wbuf[h * 4 + c]     = w;
  wbuf[h * 4 + 2 + c] = sqrtf(2.0f * w);
}

/* ---------------- build bf16 q_hat/k_hat (H,N,32) and v (H,N,32) ---------------- */
__global__ void hat_kernel(const float* __restrict__ q, const float* __restrict__ k,
                           const float* __restrict__ v, const float* __restrict__ coords,
                           const float* __restrict__ wbuf,
                           unsigned short* __restrict__ qhat,
                           unsigned short* __restrict__ khat,
                           unsigned short* __restrict__ vbf) {
  int t = blockIdx.x * blockDim.x + threadIdx.x;
  if (t >= NH * NPTS) return;
  int h = t / NPTS, n = t % NPTS;
  float w0 = wbuf[h * 4 + 0], w1 = wbuf[h * 4 + 1];
  float s0 = wbuf[h * 4 + 2], s1 = wbuf[h * 4 + 3];
  float p0 = coords[n * 3 + 1], p1 = coords[n * 3 + 2];
  float sqn = w0 * p0 * p0 + w1 * p1 * p1;
  float qp0 = s0 * p0, qp1 = s1 * p1;
  const float scale = rsqrtf((float)HD);
  size_t ro = ((size_t)h * NPTS + n) * 32;
  size_t qo = (size_t)n * 192 + h * HD;
  #pragma unroll
  for (int j = 0; j < HD; ++j) {
    qhat[ro + j] = f2bf(q[qo + j] * scale);
    khat[ro + j] = f2bf(k[qo + j]);
    vbf[ro + j]  = f2bf(v[qo + j]);
  }
  qhat[ro + 24] = f2bf(qp0); qhat[ro + 25] = f2bf(qp1);
  qhat[ro + 26] = f2bf(-sqn); qhat[ro + 27] = f2bf(1.0f);
  khat[ro + 24] = f2bf(qp0); khat[ro + 25] = f2bf(qp1);
  khat[ro + 26] = f2bf(1.0f); khat[ro + 27] = f2bf(-sqn);
  #pragma unroll
  for (int j = 28; j < 32; ++j) { qhat[ro + j] = 0; khat[ro + j] = 0; }
  #pragma unroll
  for (int j = 24; j < 32; ++j) vbf[ro + j] = 0;
}

/* ---------------- hash score + full LDS-resident bitonic argsort --------------
   One workgroup per (side, head, hash). 32768 u64 (keyBits<<32 | idx) pairs
   live entirely in 256 KB of CDNA5's 320 KB LDS.                              */
__global__ __launch_bounds__(1024) void score_sort_kernel(
    const unsigned short* __restrict__ qhat, const unsigned short* __restrict__ khat,
    const float* __restrict__ alphas,
    unsigned* __restrict__ permq, unsigned* __restrict__ permk) {
  extern __shared__ unsigned long long skey[];
  int side = blockIdx.x, h = blockIdx.y, r = blockIdx.z;
  const unsigned short* hat = side ? khat : qhat;
  unsigned* perm = side ? permk : permq;
  const float* al = alphas + (size_t)(r * NH + h) * HASHD;
  float a[HASHD];
  #pragma unroll
  for (int e = 0; e < HASHD; ++e) a[e] = al[e];

  for (int n = threadIdx.x; n < NPTS; n += 1024) {
    const unsigned short* row = hat + ((size_t)h * NPTS + n) * 32;
    float s = 0.f;
    #pragma unroll
    for (int e = 0; e < HASHD; ++e) s += bf2f(row[e]) * a[e];
    unsigned bits = __float_as_uint(s);
    bits = (bits & 0x80000000u) ? ~bits : (bits | 0x80000000u); // float -> sortable
    skey[n] = (((unsigned long long)bits) << 32) | (unsigned)n;
  }
  __syncthreads();

  for (unsigned k = 2; k <= (unsigned)NPTS; k <<= 1) {
    for (unsigned j = k >> 1; j > 0; j >>= 1) {
      for (unsigned i = threadIdx.x; i < (unsigned)NPTS; i += 1024) {
        unsigned ixj = i ^ j;
        if (ixj > i) {
          unsigned long long A = skey[i], B = skey[ixj];
          bool up = ((i & k) == 0);
          if ((A > B) == up) { skey[i] = B; skey[ixj] = A; }
        }
      }
      __syncthreads();
    }
  }

  for (int p = threadIdx.x; p < NPTS; p += 1024)
    perm[((size_t)(r * NH + h)) * NPTS + p] = (unsigned)(skey[p] & 0xFFFFFFFFu);
}

/* ---------------- block attention with WMMA -----------------------------------
   Grid (NB, H, R), 256 threads = 8 waves. Wave w owns rows [16w,16w+16) of the
   128-row block. Permuted q_hat/k_hat rows are gathered straight into LDS via
   global_load_async_to_lds_b128 (ASYNCcnt) while the VALU transposes V.
   S = q_hat @ k_hat^T: one v_wmma_f32_16x16x32_bf16 per 16x16 tile (K=28 pad
   32). Softmax+logsumexp via half-wave shfl_xor. P staged bf16 in LDS,
   O = P @ V via 4 chained K=32 WMMAs per output tile.                         */
__global__ __launch_bounds__(256) void attn_kernel(
    const unsigned short* __restrict__ qhat, const unsigned short* __restrict__ khat,
    const unsigned short* __restrict__ vbf,
    const unsigned* __restrict__ permq, const unsigned* __restrict__ permk,
    float* __restrict__ outr, float* __restrict__ zr) {
  __shared__ unsigned short qs[BLK][32];
  __shared__ unsigned short ks[BLK][32];
  __shared__ unsigned short vsT[32][BLK];
  __shared__ unsigned short pst[8][16][BLK];
  __shared__ unsigned iqp[BLK];

  const int b = blockIdx.x, h = blockIdx.y, r = blockIdx.z;
  const int t = threadIdx.x;
  const size_t permbase = ((size_t)(r * NH + h)) * NPTS + (size_t)b * BLK;

  { /* gather sorted rows: 2 threads per row, 32 B each */
    int row = t >> 1, seg = t & 1;
    unsigned nq = permq[permbase + row];
    unsigned nk = permk[permbase + row];
    if (seg == 0) iqp[row] = nq;
    const unsigned short* qsrc = qhat + ((size_t)h * NPTS + nq) * 32 + seg * 16;
    const unsigned short* ksrc = khat + ((size_t)h * NPTS + nk) * 32 + seg * 16;
    /* async global -> LDS, no VGPR staging */
    async_g2l_b128(&qs[row][seg * 16 + 0], qsrc);
    async_g2l_b128(&qs[row][seg * 16 + 8], qsrc + 8);
    async_g2l_b128(&ks[row][seg * 16 + 0], ksrc);
    async_g2l_b128(&ks[row][seg * 16 + 8], ksrc + 8);
    /* V: vectorized fetch + register transpose (overlaps with async engine) */
    const U4* vsrc4 = (const U4*)(vbf + ((size_t)h * NPTS + nk) * 32);
    union { U4 u[2]; unsigned short s[16]; } vrow;
    vrow.u[0] = vsrc4[seg * 2 + 0];
    vrow.u[1] = vsrc4[seg * 2 + 1];
    #pragma unroll
    for (int e = 0; e < 16; ++e)
      vsT[seg * 16 + e][row] = vrow.s[e];
  }
  wait_async0();
  __syncthreads();

  const int w = t >> 5, lane = t & 31;
  const int lrow = lane & 15;
  const int hi = (lane & 16) ? 1 : 0;

  union FragBF { v16bf v; U4 u[2]; };
  union FAcc   { v8f v; float f[8]; };
  v8f zero = {};

  /* ---- S = qs · ks^T : A frag (16x32) loaded once, 8 B frags ---- */
  FragBF fa;
  {
    const unsigned short* base = &qs[w * 16 + lrow][0];
    int kb = hi ? 16 : 0;                     /* A: elems 0-15 = K kb..kb+15 */
    fa.u[0] = *(const U4*)(base + kb);
    fa.u[1] = *(const U4*)(base + kb + 8);
  }
  FAcc acc[8];
  #pragma unroll
  for (int nt = 0; nt < 8; ++nt) {
    FragBF fb;
    const unsigned short* base = &ks[nt * 16 + lrow][0];
    int kb2 = hi ? 16 : 0;                    /* B: elems 0-15 = K kb2..kb2+15 */
    fb.u[0] = *(const U4*)(base + kb2);
    fb.u[1] = *(const U4*)(base + kb2 + 8);
    acc[nt].v = __builtin_amdgcn_wmma_f32_16x16x32_bf16(
        false, fa.v, false, fb.v, (short)0, zero, false, false);
  }

  /* ---- softmax + logsumexp: row m lives in one 16-lane half ---- */
  float rsum[8], zrow[8];
  #pragma unroll
  for (int rr = 0; rr < 8; ++rr) {
    float m = -3.402823466e38f;
    #pragma unroll
    for (int nt = 0; nt < 8; ++nt) m = fmaxf(m, acc[nt].f[rr]);
    m = fmaxf(m, __shfl_xor(m, 1, 32));
    m = fmaxf(m, __shfl_xor(m, 2, 32));
    m = fmaxf(m, __shfl_xor(m, 4, 32));
    m = fmaxf(m, __shfl_xor(m, 8, 32));
    float s = 0.f;
    #pragma unroll
    for (int nt = 0; nt < 8; ++nt) {
      float p = __expf(acc[nt].f[rr] - m);
      acc[nt].f[rr] = p;
      s += p;
    }
    s += __shfl_xor(s, 1, 32);
    s += __shfl_xor(s, 2, 32);
    s += __shfl_xor(s, 4, 32);
    s += __shfl_xor(s, 8, 32);
    rsum[rr] = 1.0f / s;
    zrow[rr] = m + __logf(s);
  }

  /* ---- stage normalized P as bf16 (D-layout -> row-major) ---- */
  #pragma unroll
  for (int nt = 0; nt < 8; ++nt) {
    int col = nt * 16 + lrow;
    #pragma unroll
    for (int rr = 0; rr < 8; ++rr)
      pst[w][rr + hi * 8][col] = f2bf(acc[nt].f[rr] * rsum[rr]);
  }
  __syncthreads();

  /* ---- O = P · V : K=128 as 4 chained K=32 WMMAs, 2 N-tiles ---- */
  FAcc oac[2];
  oac[0].v = zero; oac[1].v = zero;
  #pragma unroll
  for (int c = 0; c < 4; ++c) {
    FragBF pa;
    const unsigned short* base = &pst[w][lrow][c * 32];
    int kb = hi ? 16 : 0;
    pa.u[0] = *(const U4*)(base + kb);
    pa.u[1] = *(const U4*)(base + kb + 8);
    #pragma unroll
    for (int jt = 0; jt < 2; ++jt) {
      FragBF vb;
      const unsigned short* vbase = &vsT[jt * 16 + lrow][c * 32];
      int kb2 = hi ? 16 : 0;
      vb.u[0] = *(const U4*)(vbase + kb2);
      vb.u[1] = *(const U4*)(vbase + kb2 + 8);
      oac[jt].v = __builtin_amdgcn_wmma_f32_16x16x32_bf16(
          false, pa.v, false, vb.v, (short)0, oac[jt].v, false, false);
    }
  }

  /* ---- scatter through the permutation (fused inverse-argsort) ---- */
  const size_t obase = ((size_t)(r * NH + h)) * NPTS;
  #pragma unroll
  for (int rr = 0; rr < 8; ++rr) {
    int m = rr + hi * 8;
    unsigned orig = iqp[w * 16 + m];
    #pragma unroll
    for (int jt = 0; jt < 2; ++jt) {
      int j = jt * 16 + lrow;
      if (j < HD) outr[(obase + orig) * HD + j] = oac[jt].f[rr];
    }
    if (lrow == 0) zr[obase + orig] = zrow[rr];
  }
}

/* ---------------- combine 3 hashes with softmax(z) weights ---------------- */
__global__ void combine_kernel(const float* __restrict__ outr, const float* __restrict__ zr,
                               float* __restrict__ aggr) {
  int t = blockIdx.x * blockDim.x + threadIdx.x;
  if (t >= NPTS * NH * HD) return;
  int n = t / (NH * HD);
  int col = t % (NH * HD);
  int h = col / HD, j = col % HD;
  size_t i0 = ((size_t)(0 * NH + h)) * NPTS + n;
  size_t i1 = ((size_t)(1 * NH + h)) * NPTS + n;
  size_t i2 = ((size_t)(2 * NH + h)) * NPTS + n;
  float z0 = zr[i0], z1 = zr[i1], z2 = zr[i2];
  float m = fmaxf(z0, fmaxf(z1, z2));
  float e0 = __expf(z0 - m), e1 = __expf(z1 - m), e2 = __expf(z2 - m);
  float inv = 1.0f / (e0 + e1 + e2);
  float o = (e0 * outr[i0 * HD + j] + e1 * outr[i1 * HD + j] + e2 * outr[i2 * HD + j]) * inv;
  aggr[(size_t)n * (NH * HD) + col] = o;
}

/* ---------------- output projection + residual ---------------- */
__global__ void proj_kernel(const float* __restrict__ aggr, const float* __restrict__ Wo,
                            const float* __restrict__ bo, const float* __restrict__ x,
                            float* __restrict__ x1) {
  int t = blockIdx.x * blockDim.x + threadIdx.x;
  if (t >= NPTS * HD) return;
  int n = t / HD, jo = t % HD;
  const float* ar = aggr + (size_t)n * 192;
  float s = bo[jo];
  for (int c = 0; c < 192; ++c) s += ar[c] * Wo[c * HD + jo];
  x1[t] = x[t] + s;
}

/* ---------------- LN2 + FFN + residual ---------------- */
__global__ void ffn_kernel(const float* __restrict__ x1, const float* __restrict__ g,
                           const float* __restrict__ bln, const float* __restrict__ W1,
                           const float* __restrict__ b1, const float* __restrict__ W2,
                           const float* __restrict__ b2, float* __restrict__ out) {
  int n = blockIdx.x * blockDim.x + threadIdx.x;
  if (n >= NPTS) return;
  const float* r = x1 + (size_t)n * HD;
  float mu = 0.f;
  #pragma unroll
  for (int i = 0; i < HD; ++i) mu += r[i];
  mu *= (1.0f / HD);
  float var = 0.f;
  #pragma unroll
  for (int i = 0; i < HD; ++i) { float d = r[i] - mu; var += d * d; }
  var *= (1.0f / HD);
  float rs = rsqrtf(var + 1e-5f);
  float hc[HD], t1[HD];
  #pragma unroll
  for (int i = 0; i < HD; ++i) hc[i] = (r[i] - mu) * rs * g[i] + bln[i];
  #pragma unroll
  for (int j = 0; j < HD; ++j) {
    float s = b1[j];
    #pragma unroll
    for (int i = 0; i < HD; ++i) s += hc[i] * W1[i * HD + j];
    t1[j] = fmaxf(s, 0.0f);
  }
  #pragma unroll
  for (int j = 0; j < HD; ++j) {
    float s = b2[j];
    #pragma unroll
    for (int i = 0; i < HD; ++i) s += t1[i] * W2[i * HD + j];
    out[(size_t)n * HD + j] = r[j] + s;
  }
}

extern "C" void kernel_launch(void* const* d_in, const int* in_sizes, int n_in,
                              void* d_out, int out_size, void* d_ws, size_t ws_size,
                              hipStream_t stream) {
  const float* x      = (const float*)d_in[0];
  const float* coords = (const float*)d_in[1];
  const float* n1g    = (const float*)d_in[2];
  const float* n1b    = (const float*)d_in[3];
  const float* Wq     = (const float*)d_in[4];
  const float* Wk     = (const float*)d_in[5];
  const float* Wv     = (const float*)d_in[6];
  const float* wrpe   = (const float*)d_in[7];
  const float* Wo     = (const float*)d_in[8];
  const float* bo     = (const float*)d_in[9];
  const float* n2g    = (const float*)d_in[10];
  const float* n2b    = (const float*)d_in[11];
  const float* W1     = (const float*)d_in[12];
  const float* b1     = (const float*)d_in[13];
  const float* W2     = (const float*)d_in[14];
  const float* b2     = (const float*)d_in[15];
  const float* alphas = (const float*)d_in[16];
  float* out = (float*)d_out;
  (void)in_sizes; (void)n_in; (void)out_size; (void)ws_size;

  char* ws = (char*)d_ws;
  size_t off = 0;
  auto alloc = [&](size_t bytes) {
    size_t p = off;
    off += (bytes + 255) & ~(size_t)255;
    return p;
  };
  float*          xn    = (float*)(ws + alloc((size_t)NPTS * HD * 4));
  float*          qkv   = (float*)(ws + alloc((size_t)NPTS * 576 * 4)); /* reused as outr */
  unsigned short* qhat  = (unsigned short*)(ws + alloc((size_t)NH * NPTS * 32 * 2));
  unsigned short* khat  = (unsigned short*)(ws + alloc((size_t)NH * NPTS * 32 * 2));
  unsigned short* vbf   = (unsigned short*)(ws + alloc((size_t)NH * NPTS * 32 * 2));
  unsigned*       permq = (unsigned*)(ws + alloc((size_t)NR * NH * NPTS * 4));
  unsigned*       permk = (unsigned*)(ws + alloc((size_t)NR * NH * NPTS * 4));
  float*          zr    = (float*)(ws + alloc((size_t)NR * NH * NPTS * 4));
  float*          aggr  = (float*)(ws + alloc((size_t)NPTS * 192 * 4));
  float*          x1    = (float*)(ws + alloc((size_t)NPTS * HD * 4));
  float*          wbuf  = (float*)(ws + alloc(256));

  float* q = qkv;
  float* k = qkv + (size_t)NPTS * 192;
  float* v = qkv + (size_t)2 * NPTS * 192;
  float* outr = qkv;  /* alias: q/k/v fp32 dead after hat_kernel; size matches exactly */

  ln1_kernel<<<(NPTS + 255) / 256, 256, 0, stream>>>(x, n1g, n1b, xn);
  qkv_kernel<<<NPTS, 192, 0, stream>>>(xn, Wq, Wk, Wv, q, k, v);
  wrpe_kernel<<<1, 32, 0, stream>>>(wrpe, wbuf);
  hat_kernel<<<(NH * NPTS + 255) / 256, 256, 0, stream>>>(q, k, v, coords, wbuf,
                                                          qhat, khat, vbf);
  {
    dim3 g(2, NH, NR);  /* 48 sorts, each fully resident in 256 KB LDS */
    score_sort_kernel<<<g, 1024, (size_t)NPTS * 8, stream>>>(qhat, khat, alphas,
                                                             permq, permk);
  }
  {
    dim3 g(NB, NH, NR);
    attn_kernel<<<g, 256, 0, stream>>>(qhat, khat, vbf, permq, permk, outr, zr);
  }
  combine_kernel<<<(NPTS * NH * HD + 255) / 256, 256, 0, stream>>>(outr, zr, aggr);
  proj_kernel<<<(NPTS * HD + 255) / 256, 256, 0, stream>>>(aggr, Wo, bo, x, x1);
  ffn_kernel<<<(NPTS + 255) / 256, 256, 0, stream>>>(x1, n2g, n2b, W1, b1, W2, b2, out);
}
